// ThoughtsFormer_17471926960408
// MI455X (gfx1250) — matrix operand order repaired and checked by reading, not compiled, not run on hardware
//
#include <hip/hip_runtime.h>
#include <hip/hip_bf16.h>
#include <math.h>

// ---------------- model constants (from reference) ----------------
#define SLEN    2048
#define DMODEL  768
#define NHEAD   8
#define HDIM    96
#define FFDIM   2048
#define VOCAB   32000
#define NLAYER  8
#define MAXLEN_ 512
#define TMAX_   4
#define LN_EPS  1e-4f

typedef __attribute__((ext_vector_type(16))) __bf16 v16bf;
typedef __attribute__((ext_vector_type(8)))  __bf16 v8bf;
typedef __attribute__((ext_vector_type(4)))  __bf16 v4bf;
typedef __attribute__((ext_vector_type(8)))  float  v8f;

#define GEMM_FLAG_GELU    1
#define GEMM_FLAG_NANZERO 2

// ---------------- bf16 fragment loaders -------------------------------------
// A (16x32 bf16): lane half h holds contiguous K = h*16..h*16+15.
// p = &Arow[k0 + half*16]. Two 16B loads.
__device__ __forceinline__ v16bf load_a_frag(const __bf16* __restrict__ p) {
  v8bf lo = *(const v8bf*)p;
  v8bf hi = *(const v8bf*)(p + 8);
  v16bf v;
#pragma unroll
  for (int t = 0; t < 8; ++t) { v[t] = lo[t]; v[8 + t] = hi[t]; }
  return v;
}
// B (32x16 bf16): lane = column, half h holds contiguous K = h*16..h*16+15.
// One 32B load (b128 pair) per fragment.
__device__ __forceinline__ v16bf load_b_frag(const __bf16* __restrict__ p) {
  return *(const v16bf*)p;
}

// ---------------- WMMA GEMM: C[M,N] = A[M,K] * W[N,K]^T (+bias)(+epilogue) ---
// 256 threads = 8 waves; workgroup tile 128x128; wave tile 32x64 (2x4 tiles).
// M % 128 == 0, N % 128 == 0, K % 32 == 0 (holds for all GEMMs here).
__global__ __launch_bounds__(256)
void gemm_bf16_wmma(const __bf16* __restrict__ A, const __bf16* __restrict__ W,
                    const float* __restrict__ bias, const float* __restrict__ res,
                    float* __restrict__ C, __bf16* __restrict__ Cbf,
                    int M, int N, int K, int flags) {
  const int tid  = threadIdx.x;
  const int wid  = tid >> 5;
  const int lane = tid & 31;
  const int half = lane >> 4;
  const int r    = lane & 15;
  const int m0 = blockIdx.y * 128 + (wid & 3) * 32;
  const int n0 = blockIdx.x * 128 + (wid >> 2) * 64;

  v8f acc[2][4] = {};

  const __bf16* a0 = A + (size_t)(m0 + r)      * K + half * 16;
  const __bf16* a1 = A + (size_t)(m0 + 16 + r) * K + half * 16;
  const __bf16* b0 = W + (size_t)(n0 + r)      * K + half * 16;
  const __bf16* b1 = W + (size_t)(n0 + 16 + r) * K + half * 16;
  const __bf16* b2 = W + (size_t)(n0 + 32 + r) * K + half * 16;
  const __bf16* b3 = W + (size_t)(n0 + 48 + r) * K + half * 16;

  for (int k0 = 0; k0 < K; k0 += 32) {
    // speculative prefetch of the next K-slab of the weight rows
    __builtin_prefetch(b0 + k0 + 128, 0, 1);
    __builtin_prefetch(b2 + k0 + 128, 0, 1);

    v16bf af[2], bf[4];
    af[0] = load_a_frag(a0 + k0);
    af[1] = load_a_frag(a1 + k0);
    bf[0] = load_b_frag(b0 + k0);
    bf[1] = load_b_frag(b1 + k0);
    bf[2] = load_b_frag(b2 + k0);
    bf[3] = load_b_frag(b3 + k0);
#pragma unroll
    for (int mi = 0; mi < 2; ++mi)
#pragma unroll
      for (int ni = 0; ni < 4; ++ni)
        acc[mi][ni] = __builtin_amdgcn_wmma_f32_16x16x32_bf16(
            false, af[mi], false, bf[ni], (short)0, acc[mi][ni], false, false);
  }

  // C 16x16 f32 tile layout: col = lane&15, row = half*8 + vgpr
  const int rowoff = half * 8;
#pragma unroll
  for (int mi = 0; mi < 2; ++mi) {
#pragma unroll
    for (int ni = 0; ni < 4; ++ni) {
      const int ncol = n0 + ni * 16 + r;
      const float bn = bias ? bias[ncol] : 0.0f;
#pragma unroll
      for (int v = 0; v < 8; ++v) {
        const int row = m0 + mi * 16 + rowoff + v;
        const size_t idx = (size_t)row * N + ncol;
        float val = acc[mi][ni][v] + bn;
        if (flags & GEMM_FLAG_GELU)
          val = 0.5f * val * (1.0f + erff(val * 0.70710678118654752f));
        if (res) val += res[idx];
        if (flags & GEMM_FLAG_NANZERO) { if (val != val) val = 0.0f; }
        if (C)   C[idx]   = val;
        if (Cbf) Cbf[idx] = (__bf16)val;
      }
    }
  }
}

// ---------------- fp32 -> bf16 bulk convert (weights, once per launch) ------
__global__ __launch_bounds__(256)
void cvt_bf16_k(const float* __restrict__ in, __bf16* __restrict__ out, long n) {
  long i = ((long)blockIdx.x * 256 + threadIdx.x) * 4;
  if (i + 3 < n) {
    float4 f = *(const float4*)(in + i);
    v4bf o;
    o[0] = (__bf16)f.x; o[1] = (__bf16)f.y; o[2] = (__bf16)f.z; o[3] = (__bf16)f.w;
    *(v4bf*)(out + i) = o;
  }
}

// ---------------- dual positional embedding ----------------
__global__ __launch_bounds__(256)
void embed_k(const float* __restrict__ se, const float* __restrict__ pe,
             const float* __restrict__ te, const int* __restrict__ ntt,
             float* __restrict__ x) {
  int idx = blockIdx.x * 256 + threadIdx.x;
  if (idx >= SLEN * DMODEL) return;
  int T = ntt[0] + 1; T = T < 1 ? 1 : (T > TMAX_ ? TMAX_ : T);
  int i = idx / DMODEL, d = idx - i * DMODEL;
  float v = 0.0f;
  if (i < MAXLEN_ * T) {
    int blk = i / T, off = i - blk * T;
    v = se[idx] + pe[blk * DMODEL + d] + te[off * DMODEL + d];
  }
  x[idx] = v;
}

// ---------------- layer norm (one row per block), bf16 output ---------------
__global__ __launch_bounds__(256)
void layernorm_k(const float* __restrict__ x, const float* __restrict__ g,
                 const float* __restrict__ b, __bf16* __restrict__ y) {
  __shared__ float s1[8], s2[8];
  const int row = blockIdx.x, tid = threadIdx.x;
  const float* xr = x + (size_t)row * DMODEL;
  float s = 0.f, ss = 0.f;
  for (int d = tid; d < DMODEL; d += 256) { float v = xr[d]; s += v; ss += v * v; }
  for (int w = 16; w; w >>= 1) { s += __shfl_xor(s, w, 32); ss += __shfl_xor(ss, w, 32); }
  if ((tid & 31) == 0) { s1[tid >> 5] = s; s2[tid >> 5] = ss; }
  __syncthreads();
  if (tid == 0) {
    float a = 0.f, c = 0.f;
    for (int i = 0; i < 8; ++i) { a += s1[i]; c += s2[i]; }
    s1[0] = a; s2[0] = c;
  }
  __syncthreads();
  const float mean = s1[0] * (1.0f / DMODEL);
  const float var  = s2[0] * (1.0f / DMODEL) - mean * mean;
  const float inv  = rsqrtf(var + LN_EPS);
  for (int d = tid; d < DMODEL; d += 256)
    y[(size_t)row * DMODEL + d] = (__bf16)((xr[d] - mean) * inv * g[d] + b[d]);
}

// ---------------- block-sparse attention: one wave per (query, head) --------
// keys(i) = { b*T : b < tok_i } U { tok_i*T .. i }   (tok_i < MAXLEN)
__global__ __launch_bounds__(32)
void attn_k(const float* __restrict__ qkv, __bf16* __restrict__ out,
            const int* __restrict__ ntt) {
  const int i = blockIdx.x, h = blockIdx.y, lane = threadIdx.x;
  int T = ntt[0] + 1; T = T < 1 ? 1 : (T > TMAX_ ? TMAX_ : T);
  __shared__ float qs[HDIM];
  const float* qrow = qkv + (size_t)i * (3 * DMODEL) + h * HDIM;
  for (int t = lane; t < HDIM; t += 32) qs[t] = qrow[t];
  __syncthreads();

  const int tok = i / T, off = i - (i / T) * T;
  int total = 0, G = 0;
  if (tok < MAXLEN_) { G = tok; total = G + off + 1; }

  const float scale = 0.102062072615966f; // 1/sqrt(96)
  const float* Kb = qkv + DMODEL + h * HDIM;
  const float* Vb = qkv + 2 * DMODEL + h * HDIM;

  float m = -__builtin_inff(), lsum = 0.0f;
  float o0 = 0.f, o1 = 0.f, o2 = 0.f;
  const int dbase = lane * 3;
  const int nch = (total + 31) >> 5;

  for (int c0 = 0; c0 < nch; ++c0) {
    const int cidx = (c0 << 5) + lane;
    int j = 0; float s = -__builtin_inff();
    if (cidx < total) {
      j = (cidx < G) ? cidx * T : (tok * T + (cidx - G));
      const float* kr = Kb + (size_t)j * (3 * DMODEL);
      float acc = 0.f;
#pragma unroll 8
      for (int d = 0; d < HDIM; ++d) acc += qs[d] * kr[d];
      s = acc * scale;
    }
    float cm = s;
    for (int w = 16; w; w >>= 1) cm = fmaxf(cm, __shfl_xor(cm, w, 32));
    const float nm = fmaxf(m, cm);
    const float corr = __expf(m - nm);  // exp(-inf)=0 on first chunk
    const float wgt = (cidx < total) ? __expf(s - nm) : 0.0f;
    float wsum = wgt;
    for (int w = 16; w; w >>= 1) wsum += __shfl_xor(wsum, w, 32);
    lsum = lsum * corr + wsum;
    o0 *= corr; o1 *= corr; o2 *= corr;
#pragma unroll 4
    for (int jj = 0; jj < 32; ++jj) {
      const float wj = __shfl(wgt, jj, 32);
      const int   jv = __shfl(j, jj, 32);
      if (wj > 0.0f) {
        const float* vr = Vb + (size_t)jv * (3 * DMODEL) + dbase;
        o0 += wj * vr[0]; o1 += wj * vr[1]; o2 += wj * vr[2];
      }
    }
    m = nm;
  }
  const float inv = (lsum > 0.f) ? (1.0f / lsum) : 0.0f;
  __bf16* orow = out + (size_t)i * DMODEL + h * HDIM + dbase;
  orow[0] = (__bf16)(o0 * inv);
  orow[1] = (__bf16)(o1 * inv);
  orow[2] = (__bf16)(o2 * inv);
}

// ---------------- value head: dot(v1_row(bf16), w_v2) + b_v2 ----------------
__global__ __launch_bounds__(32)
void value_k(const __bf16* __restrict__ v1, const float* __restrict__ w2,
             const float* __restrict__ b2, float* __restrict__ out) {
  const int i = blockIdx.x, lane = threadIdx.x;
  const __bf16* r = v1 + (size_t)i * FFDIM;
  float s = 0.f;
  for (int d = lane; d < FFDIM; d += 32) s += (float)r[d] * w2[d];
  for (int w = 16; w; w >>= 1) s += __shfl_xor(s, w, 32);
  if (lane == 0) out[i] = s + b2[0];
}

// ---------------- host orchestration ----------------------------------------
static void launch_gemm(const __bf16* A, const __bf16* W, const float* bias,
                        const float* res, float* C, __bf16* Cbf,
                        int M, int N, int K, int flags, hipStream_t s) {
  dim3 g(N / 128, M / 128), b(256);
  gemm_bf16_wmma<<<g, b, 0, s>>>(A, W, bias, res, C, Cbf, M, N, K, flags);
}

static void launch_cvt(const float* in, __bf16* out, long n, hipStream_t s) {
  long blocks = (n / 4 + 255) / 256;
  cvt_bf16_k<<<(int)blocks, 256, 0, s>>>(in, out, n);
}

extern "C" void kernel_launch(void* const* d_in, const int* in_sizes, int n_in,
                              void* d_out, int out_size, void* d_ws, size_t ws_size,
                              hipStream_t stream) {
  (void)in_sizes; (void)n_in; (void)out_size; (void)ws_size;
  const float* se    = (const float*)d_in[0];
  // d_in[1] = padding_mask (all false) -- unused
  const float* pe    = (const float*)d_in[2];
  const float* te    = (const float*)d_in[3];
  const float* w_qkv = (const float*)d_in[4];
  const float* b_qkv = (const float*)d_in[5];
  const float* w_out = (const float*)d_in[6];
  const float* b_out = (const float*)d_in[7];
  const float* ln1g  = (const float*)d_in[8];
  const float* ln1b  = (const float*)d_in[9];
  const float* ln2g  = (const float*)d_in[10];
  const float* ln2b  = (const float*)d_in[11];
  const float* w_ff1 = (const float*)d_in[12];
  const float* b_ff1 = (const float*)d_in[13];
  const float* w_ff2 = (const float*)d_in[14];
  const float* b_ff2 = (const float*)d_in[15];
  const float* fng   = (const float*)d_in[16];
  const float* fnb   = (const float*)d_in[17];
  const float* w_pol = (const float*)d_in[18];
  const float* w_v1  = (const float*)d_in[19];
  const float* b_v1  = (const float*)d_in[20];
  const float* w_v2  = (const float*)d_in[21];
  const float* b_v2  = (const float*)d_in[22];
  const int*   ntt   = (const int*)d_in[23];

  // ---------------- workspace layout ----------------
  char* p = (char*)d_ws;
  float* X    = (float*)p;  p += (size_t)SLEN * DMODEL * 4;       // residual stream
  float* QKV  = (float*)p;  p += (size_t)SLEN * 3 * DMODEL * 4;   // qkv (fp32 for attn)
  __bf16* Ybf   = (__bf16*)p; p += (size_t)SLEN * DMODEL * 2;     // LN output
  __bf16* ATTbf = (__bf16*)p; p += (size_t)SLEN * DMODEL * 2;     // attention output
  __bf16* FFBbf = (__bf16*)p; p += (size_t)SLEN * FFDIM * 2;      // GELU(ff1) / v1
  __bf16* wqkvB = (__bf16*)p; p += (size_t)NLAYER * 3 * DMODEL * DMODEL * 2;
  __bf16* woutB = (__bf16*)p; p += (size_t)NLAYER * DMODEL * DMODEL * 2;
  __bf16* wff1B = (__bf16*)p; p += (size_t)NLAYER * FFDIM * DMODEL * 2;
  __bf16* wff2B = (__bf16*)p; p += (size_t)NLAYER * DMODEL * FFDIM * 2;
  __bf16* wpolB = (__bf16*)p; p += (size_t)VOCAB * DMODEL * 2;
  __bf16* wv1B  = (__bf16*)p; p += (size_t)FFDIM * DMODEL * 2;

  float* logits = (float*)d_out;                   // [S, V]
  float* values = logits + (size_t)SLEN * VOCAB;   // [S, 1]

  // one-time (per launch) weight conversion fp32 -> bf16
  launch_cvt(w_qkv, wqkvB, (long)NLAYER * 3 * DMODEL * DMODEL, stream);
  launch_cvt(w_out, woutB, (long)NLAYER * DMODEL * DMODEL, stream);
  launch_cvt(w_ff1, wff1B, (long)NLAYER * FFDIM * DMODEL, stream);
  launch_cvt(w_ff2, wff2B, (long)NLAYER * DMODEL * FFDIM, stream);
  launch_cvt(w_pol, wpolB, (long)VOCAB * DMODEL, stream);
  launch_cvt(w_v1,  wv1B,  (long)FFDIM * DMODEL, stream);

  // embedding + dual positional encoding
  embed_k<<<(SLEN * DMODEL + 255) / 256, 256, 0, stream>>>(se, pe, te, ntt, X);

  for (int l = 0; l < NLAYER; ++l) {
    const __bf16* wq  = wqkvB + (size_t)l * 3 * DMODEL * DMODEL;
    const float*  bq  = b_qkv + (size_t)l * 3 * DMODEL;
    const __bf16* wo  = woutB + (size_t)l * DMODEL * DMODEL;
    const float*  bo  = b_out + (size_t)l * DMODEL;
    const __bf16* wf1 = wff1B + (size_t)l * FFDIM * DMODEL;
    const float*  bf1 = b_ff1 + (size_t)l * FFDIM;
    const __bf16* wf2 = wff2B + (size_t)l * DMODEL * FFDIM;
    const float*  bf2 = b_ff2 + (size_t)l * DMODEL;

    // pre-norm attention
    layernorm_k<<<SLEN, 256, 0, stream>>>(X, ln1g + l * DMODEL, ln1b + l * DMODEL, Ybf);
    launch_gemm(Ybf, wq, bq, nullptr, QKV, nullptr, SLEN, 3 * DMODEL, DMODEL, 0, stream);
    attn_k<<<dim3(SLEN, NHEAD), 32, 0, stream>>>(QKV, ATTbf, ntt);
    launch_gemm(ATTbf, wo, bo, X, X, nullptr, SLEN, DMODEL, DMODEL, 0, stream);

    // pre-norm FFN with exact GELU; NaN->0 at layer end
    layernorm_k<<<SLEN, 256, 0, stream>>>(X, ln2g + l * DMODEL, ln2b + l * DMODEL, Ybf);
    launch_gemm(Ybf, wf1, bf1, nullptr, nullptr, FFBbf, SLEN, FFDIM, DMODEL,
                GEMM_FLAG_GELU, stream);
    launch_gemm(FFBbf, wf2, bf2, X, X, nullptr, SLEN, DMODEL, FFDIM,
                GEMM_FLAG_NANZERO, stream);
  }

  // final norm + heads
  layernorm_k<<<SLEN, 256, 0, stream>>>(X, fng, fnb, Ybf);
  launch_gemm(Ybf, wpolB, nullptr, nullptr, logits, nullptr, SLEN, VOCAB, DMODEL, 0, stream);
  launch_gemm(Ybf, wv1B, b_v1, nullptr, nullptr, FFBbf, SLEN, FFDIM, DMODEL,
              GEMM_FLAG_GELU, stream);
  value_k<<<SLEN, 32, 0, stream>>>(FFBbf, w_v2, b_v2, values);
}
